// Attention_Net_34196529611457
// MI455X (gfx1250) — 16-bit and 8-bit matrix operand order repaired and checked by reading, compiled, not run on hardware
//
#include <hip/hip_runtime.h>
#include <hip/hip_bf16.h>
#include <math.h>


typedef __attribute__((ext_vector_type(16))) __bf16 v16bf;
typedef __attribute__((ext_vector_type(8)))  float  v8f;
typedef __attribute__((ext_vector_type(8)))  int    v8i;

#define B_   64
#define L_   1024
#define P_   64
#define H_   256
#define G_   768
#define OUT_ 128
#define NWAVES 24
#define NT_ 768


__device__ __forceinline__ unsigned char f32_to_e4m3(float f) {
  union { float f; unsigned u; } v; v.f = f;
  const unsigned u = v.u;
  const unsigned sign = (u >> 24) & 0x80u;
  const unsigned a = u & 0x7fffffffu;
  const unsigned r = a + 0x7ffffu + ((a >> 20) & 1u);
  const int ebits = (int)(r >> 23);
  int val = (int)(r >> 20) - 960;
  const int dn = (int)rintf(__builtin_fabsf(f) * 512.f);
  val = (ebits < 121) ? dn : val;
  val = (val > 0x7e) ? 0x7e : val;
  return (unsigned char)(sign | (unsigned)val);
}


__device__ __forceinline__ v8f wmma8(v8i a, v8i b, v8f c) {
  return __builtin_amdgcn_wmma_f32_16x16x64_fp8_fp8(a, b, (short)0, c, false, false);
}
__device__ __forceinline__ v8f wmma_bf16(v16bf a, v16bf b, v8f c) {
  return __builtin_amdgcn_wmma_f32_16x16x32_bf16(false, a, false, b, (short)0, c,
                                                 false, false);
}


__device__ __forceinline__ v8i load_afrag8(const unsigned char* M, int ld, int mbase, int kbase) {
  const int lane = threadIdx.x & 31;
  const int m  = mbase + (lane & 15);
  const int kb = (lane >> 4) << 3;
  const unsigned char* p = M + (size_t)m * ld + kbase + kb;
  union { v8i v; uint2 d[4]; } t;
  t.d[0] = *(const uint2*)(p);
  t.d[1] = *(const uint2*)(p + 16);
  t.d[2] = *(const uint2*)(p + 32);
  t.d[3] = *(const uint2*)(p + 48);
  return t.v;
}

__device__ __forceinline__ v8i load_bfrag8f(const unsigned char* Wf, int KF, int nt, int kf) {
  const int lane = threadIdx.x & 31;
  const unsigned char* p = Wf + (((size_t)(nt * KF + kf)) << 10) + (lane << 5);
  union { v8i v; uint4 q[2]; } t;
  t.q[0] = *(const uint4*)(p);
  t.q[1] = *(const uint4*)(p + 16);
  return t.v;
}

__device__ __forceinline__ v16bf load_bfrag16(const __bf16* Wt, int ld, int kbase, int nbase) {
  const int lane = threadIdx.x & 31;
  const int kb = (lane >> 4) << 3;
  const __bf16* p = Wt + (size_t)(kbase + kb) * ld + nbase + (lane & 15);
  v16bf r;
#pragma unroll
  for (int i = 0; i < 8; ++i) r[i] = p[(size_t)i * ld];
#pragma unroll
  for (int i = 0; i < 8; ++i) r[8 + i] = p[(size_t)(16 + i) * ld];
  return r;
}

__device__ __forceinline__ v16bf load_afrag16_f32(const float* M, int ld, int mbase, int kbase) {
  const int lane = threadIdx.x & 31;
  const int m  = mbase + (lane & 15);
  const int kb = (lane >> 4) << 3;
  const float* p = M + (size_t)m * ld + kbase + kb;
  v16bf r;
#pragma unroll
  for (int i = 0; i < 8; ++i) r[i] = (__bf16)p[i];
#pragma unroll
  for (int i = 0; i < 8; ++i) r[8 + i] = (__bf16)p[16 + i];
  return r;
}

__device__ __forceinline__ void store_cfrag_bf(__bf16* Mdst, int ld, int mbase, int nbase, v8f acc) {
  const int lane = threadIdx.x & 31;
  const int n  = nbase + (lane & 15);
  const int mo = (lane >> 4) << 3;
#pragma unroll
  for (int v = 0; v < 8; ++v)
    Mdst[(size_t)(mbase + mo + v) * ld + n] = (__bf16)acc[v];
}

__device__ __forceinline__ float sigmoidf_(float x) { return 1.f / (1.f + __expf(-x)); }

__device__ __forceinline__ v8f bias_acc(float b) {
  v8f a;
#pragma unroll
  for (int v = 0; v < 8; ++v) a[v] = b;
  return a;
}

__device__ __forceinline__ void gate_gemm_fp8(const unsigned char* A, int lda,
                                              const unsigned char* __restrict__ Wf, int K,
                                              float bias0, float bias1,
                                              int nt0, __bf16* dstG) {
  const int KF  = K >> 6;
  const int nb0 = nt0 * 16, nb1 = nb0 + 16;
  for (int mg = 0; mg < 4; mg += 2) {
    unsigned wo = 0;
    asm volatile("" : "+s"(wo));
    const unsigned char* W = Wf + wo;
    v8f c00 = bias_acc(bias0), c10 = bias_acc(bias1);
    v8f c01 = bias_acc(bias0), c11 = bias_acc(bias1);
    for (int kf = 0; kf < KF; ++kf) {
      v8i b0 = load_bfrag8f(W, KF, nt0,     kf);
      v8i b1 = load_bfrag8f(W, KF, nt0 + 1, kf);
      v8i aA = load_afrag8(A, lda, (mg + 0) * 16, kf * 64);
      v8i aB = load_afrag8(A, lda, (mg + 1) * 16, kf * 64);
      c00 = wmma8(aA, b0, c00);
      c10 = wmma8(aA, b1, c10);
      c01 = wmma8(aB, b0, c01);
      c11 = wmma8(aB, b1, c11);
    }
    store_cfrag_bf(dstG, G_, (mg + 0) * 16, nb0, c00);
    store_cfrag_bf(dstG, G_, (mg + 0) * 16, nb1, c10);
    store_cfrag_bf(dstG, G_, (mg + 1) * 16, nb0, c01);
    store_cfrag_bf(dstG, G_, (mg + 1) * 16, nb1, c11);
  }
}


__global__ void pack_fp8_frag_kernel(const float* __restrict__ W, unsigned char* __restrict__ Wf,
                                     int K) {
  int idx = blockIdx.x * blockDim.x + threadIdx.x;
  if (idx >= G_ * K) return;
  const int nt  = idx / (K * 16);
  const int rem = idx - nt * (K * 16);
  const int kf  = rem >> 10;
  const int r2  = rem & 1023;
  const int l   = r2 >> 5;
  const int j   = r2 & 31;
  const int k   = kf * 64 + ((j >> 3) << 4) + ((l >> 4) << 3) + (j & 7);
  const int n   = nt * 16 + (l & 15);
  Wf[idx] = f32_to_e4m3(W[(size_t)n * K + k]);
}

__global__ void pack_bf16_kernel(const float* __restrict__ W, __bf16* __restrict__ Wt,
                                 int rows, int cols) {
  int idx = blockIdx.x * blockDim.x + threadIdx.x;
  if (idx >= rows * cols) return;
  int n = idx / cols, k = idx - n * cols;
  Wt[(size_t)k * rows + n] = (__bf16)W[idx];
}


__global__ void __launch_bounds__(768, 1)
encoder_kernel(const float* __restrict__ x,
               const float* __restrict__ b_ih, const float* __restrict__ b_hh,
               const unsigned char* __restrict__ Wih8, const unsigned char* __restrict__ Whh8,
               __bf16* __restrict__ hseq, float* __restrict__ s_out) {
  extern __shared__ char smem[];
  unsigned char* lds_x8 = (unsigned char*)smem;
  float*         lds_h  = (float*)(smem + 4096);
  unsigned char* lds_h8 = (unsigned char*)(smem + 4096 + 65536);
  __bf16*        lds_gi = (__bf16*)(smem + 4096 + 65536 + 16384);
  __bf16*        lds_gh = (__bf16*)(smem + 4096 + 65536 + 16384 + 98304);

  const int tid  = threadIdx.x;
  const int wave = tid >> 5;
  const int lane = tid & 31;
  const int nt0  = wave * 2;

  const float bi0 = b_ih[nt0 * 16 + (lane & 15)], bi1 = b_ih[nt0 * 16 + 16 + (lane & 15)];
  const float bh0 = b_hh[nt0 * 16 + (lane & 15)], bh1 = b_hh[nt0 * 16 + 16 + (lane & 15)];

  for (int i = tid; i < B_ * H_; i += NT_) { lds_h[i] = 0.f; lds_h8[i] = 0; }
  __syncthreads();

  for (int t = 0; t < L_; ++t) {
    for (int i = tid; i < B_ * P_; i += NT_) {
      int b = i >> 6, p = i & 63;
      lds_x8[i] = f32_to_e4m3(x[((size_t)b * L_ + t) * P_ + p]);
    }
    __syncthreads();

    gate_gemm_fp8(lds_x8, P_, Wih8, 64,  bi0, bi1, nt0, lds_gi);
    gate_gemm_fp8(lds_h8, H_, Whh8, 256, bh0, bh1, nt0, lds_gh);
    __syncthreads();

    for (int i = tid; i < B_ * H_; i += NT_) {
      int b = i >> 8, j = i & 255;
      const __bf16* gip = lds_gi + (size_t)b * G_;
      const __bf16* ghp = lds_gh + (size_t)b * G_;
      float ir = (float)gip[j], iz = (float)gip[j + H_], inn = (float)gip[j + 2 * H_];
      float hr = (float)ghp[j], hz = (float)ghp[j + H_], hn  = (float)ghp[j + 2 * H_];
      float r = sigmoidf_(ir + hr);
      float z = sigmoidf_(iz + hz);
      float n = tanhf(inn + r * hn);
      float hnew = (1.f - z) * n + z * lds_h[i];
      lds_h[i]  = hnew;
      lds_h8[i] = f32_to_e4m3(hnew);
      hseq[((size_t)b * L_ + t) * H_ + j] = (__bf16)hnew;
    }
    __syncthreads();
  }

  for (int i = tid; i < B_ * H_; i += NT_) s_out[i] = lds_h[i];
}


__global__ void hproj_kernel(const __bf16* __restrict__ hseq, const float* __restrict__ W_attn,
                             float* __restrict__ hproj) {
  int idx = blockIdx.x * blockDim.x + threadIdx.x;
  if (idx >= B_ * L_) return;
  const __bf16* hp = hseq + (size_t)idx * H_;
  float acc = 0.f;
  for (int h = 0; h < H_; ++h) acc += (float)hp[h] * W_attn[H_ + h];
  hproj[idx] = acc;
}


__global__ void __launch_bounds__(768, 1)
decoder_kernel(const float* __restrict__ b_ih, const float* __restrict__ b_hh,
               const unsigned char* __restrict__ Wih8, const unsigned char* __restrict__ Whh8,
               const float* __restrict__ W_attn, const float* __restrict__ b_attn,
               const float* __restrict__ W_dec_out, const float* __restrict__ b_dec_out,
               const __bf16* __restrict__ hseq, const float* __restrict__ hproj,
               const float* __restrict__ s_init, float* __restrict__ y) {
  extern __shared__ char smem[];
  __bf16*        lds_alpha = (__bf16*)smem;
  __bf16*        lds_gi    = (__bf16*)smem;
  __bf16*        lds_gh    = (__bf16*)(smem + 98304);
  float*         lds_s     = (float*)(smem + 196608);
  unsigned char* lds_s8    = (unsigned char*)(smem + 196608 + 65536);
  unsigned char* lds_c8    = (unsigned char*)(smem + 196608 + 65536 + 16384);
  float*         lds_sdot  = (float*)(smem + 196608 + 65536 + 32768);
  float*         lds_rsum  = (float*)(smem + 196608 + 65536 + 32768 + 256);

  const int tid  = threadIdx.x;
  const int wave = tid >> 5;
  const int lane = tid & 31;
  const float ba = b_attn[0];
  const int nt0  = wave * 2;

  const float bi0 = b_ih[nt0 * 16 + (lane & 15)], bi1 = b_ih[nt0 * 16 + 16 + (lane & 15)];
  const float bh0 = b_hh[nt0 * 16 + (lane & 15)], bh1 = b_hh[nt0 * 16 + 16 + (lane & 15)];

  for (int i = tid; i < B_ * H_; i += NT_) {
    float s0 = s_init[i];
    lds_s[i]  = s0;
    lds_s8[i] = f32_to_e4m3(s0);
  }
  __syncthreads();

  for (int t = 0; t < L_; ++t) {
    if (tid < B_) {
      const float* sp = lds_s + tid * H_;
      float acc = 0.f;
      for (int h = 0; h < H_; ++h) acc += sp[h] * W_attn[h];
      lds_sdot[tid] = acc;
    }
    __syncthreads();

    for (int r = wave; r < B_; r += NWAVES) {
      const float  sd = lds_sdot[r] + ba;
      const float* hp = hproj + (size_t)r * L_;
      float mx = -3.4e38f;
      for (int l = lane; l < L_; l += 32) mx = fmaxf(mx, sd + hp[l]);
#pragma unroll
      for (int off = 16; off > 0; off >>= 1) mx = fmaxf(mx, __shfl_xor(mx, off, 32));
      float sum = 0.f;
      for (int l = lane; l < L_; l += 32) {
        float p = __expf(sd + hp[l] - mx);
        sum += p;
        lds_alpha[(size_t)r * L_ + l] = (__bf16)p;
      }
#pragma unroll
      for (int off = 16; off > 0; off >>= 1) sum += __shfl_xor(sum, off, 32);
      if (lane == 0) lds_rsum[r] = 1.f / sum;
    }
    __syncthreads();

    for (int i = tid; i < B_ * H_; i += NT_) {
      int b = i >> 8, h = i & 255;
      const __bf16* ap = lds_alpha + (size_t)b * L_;
      const __bf16* hp = hseq + (size_t)b * L_ * H_ + h;
      float acc = 0.f;
      for (int l = 0; l < L_; ++l) {
        if ((l & 63) == 0) __builtin_prefetch(hp + (size_t)(l + 64) * H_, 0, 1);
        acc += (float)ap[l] * (float)hp[(size_t)l * H_];
      }
      lds_c8[i] = f32_to_e4m3(acc * lds_rsum[b]);
    }
    __syncthreads();

    gate_gemm_fp8(lds_c8, H_, Wih8, 256, bi0, bi1, nt0, lds_gi);
    gate_gemm_fp8(lds_s8, H_, Whh8, 256, bh0, bh1, nt0, lds_gh);
    __syncthreads();

    for (int i = tid; i < B_ * H_; i += NT_) {
      int b = i >> 8, j = i & 255;
      const __bf16* gip = lds_gi + (size_t)b * G_;
      const __bf16* ghp = lds_gh + (size_t)b * G_;
      float ir = (float)gip[j], iz = (float)gip[j + H_], inn = (float)gip[j + 2 * H_];
      float hr = (float)ghp[j], hz = (float)ghp[j + H_], hn  = (float)ghp[j + 2 * H_];
      float r = sigmoidf_(ir + hr);
      float z = sigmoidf_(iz + hz);
      float n = tanhf(inn + r * hn);
      float snew = (1.f - z) * n + z * lds_s[i];
      lds_s[i]  = snew;
      lds_s8[i] = f32_to_e4m3(snew);
    }
    __syncthreads();

    if (tid < B_) {
      const float* sp = lds_s + tid * H_;
      float acc = b_dec_out[0];
      for (int h = 0; h < H_; ++h) acc += sp[h] * W_dec_out[h];
      y[(size_t)tid * L_ + t] = sigmoidf_(acc);
    }
    __syncthreads();
  }
}


__global__ void __launch_bounds__(256, 1)
outgemm_kernel(const float* __restrict__ y, const __bf16* __restrict__ Wout_t  ,
               const float* __restrict__ b_out, float* __restrict__ out) {
  const int wave = threadIdx.x >> 5;
  const int lane = threadIdx.x & 31;
  for (int tile = wave; tile < 32; tile += 8) {
    const int mt = tile >> 3, nt = tile & 7;
    const int nb = nt * 16;
    v8f acc = bias_acc(b_out[nb + (lane & 15)]);
    for (int kf = 0; kf < 32; ++kf) {
      v16bf a = load_afrag16_f32(y, L_, mt * 16, kf * 32);
      v16bf b = load_bfrag16(Wout_t, OUT_, kf * 32, nb);
      acc = wmma_bf16(a, b, acc);
    }
    const int n  = nb + (lane & 15);
    const int mo = (lane >> 4) << 3;
#pragma unroll
    for (int v = 0; v < 8; ++v)
      out[(size_t)(mt * 16 + mo + v) * OUT_ + n] = acc[v];
  }
}


extern "C" void kernel_launch(void* const* d_in, const int* in_sizes, int n_in,
                              void* d_out, int out_size, void* d_ws, size_t ws_size,
                              hipStream_t stream) {
  (void)in_sizes; (void)n_in; (void)out_size; (void)ws_size;
  const float* x         = (const float*)d_in[0];
  const float* W_ih_e    = (const float*)d_in[1];
  const float* W_hh_e    = (const float*)d_in[2];
  const float* b_ih_e    = (const float*)d_in[3];
  const float* b_hh_e    = (const float*)d_in[4];
  const float* W_ih_d    = (const float*)d_in[5];
  const float* W_hh_d    = (const float*)d_in[6];
  const float* b_ih_d    = (const float*)d_in[7];
  const float* b_hh_d    = (const float*)d_in[8];
  const float* W_dec_out = (const float*)d_in[9];
  const float* b_dec_out = (const float*)d_in[10];
  const float* W_attn    = (const float*)d_in[11];
  const float* b_attn    = (const float*)d_in[12];
  const float* W_out     = (const float*)d_in[13];
  const float* b_out     = (const float*)d_in[14];

  char* ws = (char*)d_ws;
  size_t off = 0;
  auto alloc = [&](size_t bytes) -> char* {
    char* p = ws + off;
    off = (off + bytes + 255) & ~(size_t)255;
    return p;
  };
  unsigned char* Wih_e8 = (unsigned char*)alloc((size_t)G_ * P_);
  unsigned char* Whh_e8 = (unsigned char*)alloc((size_t)G_ * H_);
  unsigned char* Wih_d8 = (unsigned char*)alloc((size_t)G_ * H_);
  unsigned char* Whh_d8 = (unsigned char*)alloc((size_t)G_ * H_);
  __bf16*        Wout_t = (__bf16*)alloc((size_t)L_ * OUT_ * 2);
  __bf16*        hseq   = (__bf16*)alloc((size_t)B_ * L_ * H_ * 2);
  float*         hproj  = (float*)alloc((size_t)B_ * L_ * 4);
  float*         yb     = (float*)alloc((size_t)B_ * L_ * 4);
  float*         s_ws   = (float*)alloc((size_t)B_ * H_ * 4);

  pack_fp8_frag_kernel<<<(G_ * P_ + 255) / 256, 256, 0, stream>>>(W_ih_e, Wih_e8, P_);
  pack_fp8_frag_kernel<<<(G_ * H_ + 255) / 256, 256, 0, stream>>>(W_hh_e, Whh_e8, H_);
  pack_fp8_frag_kernel<<<(G_ * H_ + 255) / 256, 256, 0, stream>>>(W_ih_d, Wih_d8, H_);
  pack_fp8_frag_kernel<<<(G_ * H_ + 255) / 256, 256, 0, stream>>>(W_hh_d, Whh_d8, H_);
  pack_bf16_kernel<<<(OUT_ * L_ + 255) / 256, 256, 0, stream>>>(W_out, Wout_t, OUT_, L_);

  const size_t enc_lds = 4096 + 65536 + 16384 + 98304 + 98304;
  hipFuncSetAttribute(reinterpret_cast<const void*>(encoder_kernel),
                      hipFuncAttributeMaxDynamicSharedMemorySize, (int)enc_lds);
  encoder_kernel<<<1, 768, enc_lds, stream>>>(x, b_ih_e, b_hh_e, Wih_e8, Whh_e8, hseq, s_ws);

  hproj_kernel<<<(B_ * L_ + 255) / 256, 256, 0, stream>>>(hseq, W_attn, hproj);

  const size_t dec_lds = 196608 + 65536 + 32768 + 256 + 256;
  hipFuncSetAttribute(reinterpret_cast<const void*>(decoder_kernel),
                      hipFuncAttributeMaxDynamicSharedMemorySize, (int)dec_lds);
  decoder_kernel<<<1, 768, dec_lds, stream>>>(b_ih_d, b_hh_d, Wih_d8, Whh_d8,
                                              W_attn, b_attn, W_dec_out, b_dec_out,
                                              hseq, hproj, s_ws, yb);

  outgemm_kernel<<<1, 256, 0, stream>>>(yb, Wout_t, b_out, (float*)d_out);
}
